// MultiHeadAttention2D_10685878632499
// MI455X (gfx1250) — matrix operand order repaired and checked by reading, not compiled, not run on hardware
//
#include <hip/hip_runtime.h>
#include <math.h>

#define NN 4096   // H*W
#define CC 256    // channels
#define CQ 32     // head dim

typedef __attribute__((ext_vector_type(16))) __bf16 v16bf;
typedef __attribute__((ext_vector_type(8)))  float  v8f;

struct I4x2 { int4 lo, hi; };

// Build a v16bf A/B operand from two 16-byte runs.
__device__ __forceinline__ v16bf ld_pair(const __bf16* p0, const __bf16* p1) {
  I4x2 t;
  t.lo = *reinterpret_cast<const int4*>(p0);
  t.hi = *reinterpret_cast<const int4*>(p1);
  return __builtin_bit_cast(v16bf, t);
}

// ---------------- projection: pqT / pkT (bf16, [B][N][CQ]) ----------------
__global__ __launch_bounds__(256)
void proj_qk_kernel(const float* __restrict__ query, const float* __restrict__ key_,
                    const float* __restrict__ Wq, const float* __restrict__ bq,
                    const float* __restrict__ Wk, const float* __restrict__ bk,
                    __bf16* __restrict__ pqT, __bf16* __restrict__ pkT)
{
  __shared__ float Wl[32 * 257];          // padded to kill bank conflicts
  const int sel = blockIdx.y;             // 0 = Q (pre-scaled by 1/64), 1 = K
  const int b   = blockIdx.z;
  const int n0  = blockIdx.x * 8;
  const float* W    = sel ? Wk  : Wq;
  const float* bias = sel ? bk  : bq;
  const float* X    = sel ? key_ : query;
  __bf16* outp      = sel ? pkT : pqT;
  const float scale = sel ? 1.0f : (1.0f / 64.0f);   // 1/sqrt(N) folded into Q

  for (int i = threadIdx.x; i < 32 * 256; i += 256)
    Wl[(i >> 8) * 257 + (i & 255)] = W[i];
  __syncthreads();

  const int cq = threadIdx.x & 31;
  const int n  = n0 + (threadIdx.x >> 5);
  const float* xcol = X + (((size_t)b * CC) << 12) + n;
  const float* wrow = &Wl[cq * 257];
  float acc = 0.f;
  for (int c = 0; c < CC; ++c)
    acc += wrow[c] * xcol[(size_t)c << 12];
  acc = (acc + bias[cq]) * scale;
  outp[((size_t)(b * NN + n)) * CQ + cq] = (__bf16)acc;
}

// ---------------- projection: pv (bf16, [B][C][N]) ----------------
__global__ __launch_bounds__(256)
void proj_v_kernel(const float* __restrict__ v,
                   const float* __restrict__ Wv, const float* __restrict__ bv,
                   __bf16* __restrict__ pvB)
{
  const int b   = blockIdx.z;
  const int cob = blockIdx.y;                     // 16 channels per block
  const int n   = blockIdx.x * 256 + threadIdx.x;
  float acc[16];
#pragma unroll
  for (int i = 0; i < 16; ++i) acc[i] = 0.f;
  const float* vcol = v + (((size_t)b * CC) << 12) + n;
  for (int c = 0; c < CC; ++c) {
    float val = vcol[(size_t)c << 12];            // coalesced per wave
#pragma unroll
    for (int i = 0; i < 16; ++i)                  // Wv indices uniform -> s_load
      acc[i] += Wv[(cob * 16 + i) * CC + c] * val;
  }
#pragma unroll
  for (int i = 0; i < 16; ++i) {
    int co = cob * 16 + i;
    pvB[((size_t)(b * CC + co) << 12) + n] = (__bf16)(acc[i] + bv[co]);
  }
}

// ---------------- fused flash attention ----------------
// One wave per (b, 16-query tile, 128-channel group).
// 64 keys per iteration: 4 S-WMMAs, one softmax pass, 16 O-WMMAs, 2 rowsum-WMMAs.
__global__ __launch_bounds__(32)
void attn_kernel(const __bf16* __restrict__ pqT,
                 const __bf16* __restrict__ pkT,
                 const __bf16* __restrict__ pvB,
                 const float*  __restrict__ value,
                 float* __restrict__ out)
{
  __shared__ __align__(64) __bf16 Pl[16 * 64];    // P tile staging (C-layout -> A-layout)
  const int lane = threadIdx.x;
  const int l16  = lane & 15;
  const int hi16 = lane >> 4;                     // lane half selects K-range / M-range
  const int b  = blockIdx.z;
  const int qt = blockIdx.x;
  const int cg = blockIdx.y;
  const int cbase = cg * 128;

  // Q tile as WMMA A operand (16 queries x 32 dims), rows from pqT[n][c]
  const __bf16* qrow = pqT + ((size_t)(b * NN + qt * 16 + l16)) * CQ;
  v16bf Qa = ld_pair(qrow + hi16 * 8, qrow + 16 + hi16 * 8);

  const __bf16 one = (__bf16)1.0f;
  const v16bf OnesB = {one, one, one, one, one, one, one, one,
                       one, one, one, one, one, one, one, one};

  v8f zero = {0.f, 0.f, 0.f, 0.f, 0.f, 0.f, 0.f, 0.f};
  v8f o[8];
#pragma unroll
  for (int t = 0; t < 8; ++t) o[t] = zero;
  v8f osum = zero;                                // rowsum accumulator (P x Ones)
  float rm[8];
#pragma unroll
  for (int j = 0; j < 8; ++j) rm[j] = -INFINITY;

  const __bf16* pkbase = pkT + (size_t)b * NN * CQ;
  const __bf16* pvbase = pvB + (size_t)b * CC * NN;

  for (int m0 = 0; m0 < NN; m0 += 64) {
    // ---- S = Q x K^T over 64 keys: 4 subtiles of 16 ----
    v8f s[4];
#pragma unroll
    for (int u = 0; u < 4; ++u) {
      const __bf16* kr = pkbase + (size_t)(m0 + u * 16 + l16) * CQ;
      v16bf kb = ld_pair(kr + hi16 * 8, kr + 16 + hi16 * 8);
      s[u] = __builtin_amdgcn_wmma_f32_16x16x32_bf16(false, Qa, false, kb,
                                                     (short)0, zero, false, false);
    }
    if (m0 + 64 < NN)
      __builtin_prefetch(pkbase + (size_t)(m0 + 64 + l16) * CQ, 0, 3);

    // ---- online softmax: one max-reduction per row per 64 keys ----
#pragma unroll
    for (int j = 0; j < 8; ++j) {
      float a = fmaxf(fmaxf(s[0][j], s[1][j]), fmaxf(s[2][j], s[3][j]));
      a = fmaxf(a, __shfl_xor(a, 1, 16));
      a = fmaxf(a, __shfl_xor(a, 2, 16));
      a = fmaxf(a, __shfl_xor(a, 4, 16));
      a = fmaxf(a, __shfl_xor(a, 8, 16));
      float nm   = fmaxf(rm[j], a);
      float corr = __expf(rm[j] - nm);
      rm[j] = nm;
      int q = j + hi16 * 8;
#pragma unroll
      for (int u = 0; u < 4; ++u) {
        float p = __expf(s[u][j] - nm);
        Pl[q * 64 + u * 16 + l16] = (__bf16)p;
      }
      // rescale previous accumulators (row j of every tile lives in VGPR j)
      osum[j] *= corr;
#pragma unroll
      for (int t = 0; t < 8; ++t) o[t][j] *= corr;
    }
    __syncthreads();   // single-wave WG: barrier is a NOP, acts as compiler fence

    // ---- re-read P in A-operand layout: two 16x32 tiles ----
    const __bf16* prow = &Pl[l16 * 64];
    v16bf Pa0 = ld_pair(prow + hi16 * 16,      prow + 8 + hi16 * 16);
    v16bf Pa1 = ld_pair(prow + 32 + hi16 * 16, prow + 40 + hi16 * 16);

    // rowsum via WMMA against all-ones B (replaces shuffle-sum tree)
    osum = __builtin_amdgcn_wmma_f32_16x16x32_bf16(false, Pa0, false, OnesB,
                                                   (short)0, osum, false, false);
    osum = __builtin_amdgcn_wmma_f32_16x16x32_bf16(false, Pa1, false, OnesB,
                                                   (short)0, osum, false, false);

    // ---- O += P x V^T for 8 channel tiles ----
#pragma unroll
    for (int t = 0; t < 8; ++t) {
      const __bf16* vr = pvbase + (size_t)(cbase + t * 16 + l16) * NN + m0 + hi16 * 16;
      v16bf vb0 = ld_pair(vr,      vr + 8);
      v16bf vb1 = ld_pair(vr + 32, vr + 40);
      o[t] = __builtin_amdgcn_wmma_f32_16x16x32_bf16(false, Pa0, false, vb0,
                                                     (short)0, o[t], false, false);
      o[t] = __builtin_amdgcn_wmma_f32_16x16x32_bf16(false, Pa1, false, vb1,
                                                     (short)0, o[t], false, false);
    }
    __syncthreads();
  }

  // Epilogue: normalize (rowsum replicated in every lane), add residual, store fp32
  float inv[8];
#pragma unroll
  for (int j = 0; j < 8; ++j) inv[j] = 1.0f / osum[j];
#pragma unroll
  for (int t = 0; t < 8; ++t) {
    int c = cbase + t * 16 + l16;
#pragma unroll
    for (int j = 0; j < 8; ++j) {
      int n = qt * 16 + j + hi16 * 8;
      size_t idx = ((size_t)(b * CC + c) << 12) + n;
      out[idx] = o[t][j] * inv[j] + value[idx];
    }
  }
}

extern "C" void kernel_launch(void* const* d_in, const int* in_sizes, int n_in,
                              void* d_out, int out_size, void* d_ws, size_t ws_size,
                              hipStream_t stream)
{
  const float* query = (const float*)d_in[0];
  const float* key_  = (const float*)d_in[1];
  const float* value = (const float*)d_in[2];
  const float* Wq    = (const float*)d_in[3];
  const float* bq    = (const float*)d_in[4];
  const float* Wk    = (const float*)d_in[5];
  const float* bk    = (const float*)d_in[6];
  const float* Wv    = (const float*)d_in[7];
  const float* bv    = (const float*)d_in[8];
  float* out = (float*)d_out;

  // workspace: pqT 1MB | pkT 1MB | pvB 8MB  (all bf16)
  __bf16* pqT = (__bf16*)d_ws;
  __bf16* pkT = (__bf16*)((char*)d_ws + (1u << 20));
  __bf16* pvB = (__bf16*)((char*)d_ws + (2u << 20));

  proj_qk_kernel<<<dim3(NN / 8, 2, 4), 256, 0, stream>>>(query, key_, Wq, bq, Wk, bk, pqT, pkT);
  proj_v_kernel<<<dim3(NN / 256, CC / 16, 4), 256, 0, stream>>>(value, Wv, bv, pvB);
  attn_kernel<<<dim3(NN / 16, 2, 4), 32, 0, stream>>>(pqT, pkT, pvB, value, out);
}
